// GCN_48661979464283
// MI455X (gfx1250) — matrix operand order repaired and checked by reading, not compiled, not run on hardware
//
#include <hip/hip_runtime.h>
#include <hip/hip_bf16.h>

typedef __attribute__((ext_vector_type(16))) _Float16 v16h;
typedef __attribute__((ext_vector_type(8)))  _Float16 v8h;
typedef __attribute__((ext_vector_type(8)))  float    v8f;
typedef __attribute__((ext_vector_type(4)))  float    f4;

#define NND   100000
#define NEDG  3200000
#define DIN   64
#define DED   16
#define HD    16

__device__ __forceinline__ v16h combine8(v8h lo, v8h hi) {
  return __builtin_shufflevector(lo, hi, 0,1,2,3,4,5,6,7,8,9,10,11,12,13,14,15);
}
__device__ __forceinline__ v8h load8h(const _Float16* p) {
  return *(const v8h*)p;  // 16B aligned -> global_load_b128
}
// 8 fp32 (non-temporal stream) -> 8 f16
__device__ __forceinline__ v8h cvt8_nt(const float* p) {
  f4 a = __builtin_nontemporal_load((const f4*)p);
  f4 b = __builtin_nontemporal_load((const f4*)(p + 4));
  v8h r;
#pragma unroll
  for (int i = 0; i < 4; ++i) { r[i] = (_Float16)a[i]; r[i + 4] = (_Float16)b[i]; }
  return r;
}
__device__ __forceinline__ v8f wmma16(v16h a, v16h b, v8f c) {
  return __builtin_amdgcn_wmma_f32_16x16x32_f16(false, a, false, b, (short)0, c, false, false);
}

// ---------------- prep kernels ----------------
__global__ void k_zero(float* p, int n) {
  int i = blockIdx.x * blockDim.x + threadIdx.x;
  if (i < n) p[i] = 0.f;
}

__global__ void k_cvt(const float* __restrict__ src, _Float16* __restrict__ dst, int n) {
  int i = blockIdx.x * blockDim.x + threadIdx.x;
  if (i < n) dst[i] = (_Float16)src[i];
}

// Pack W (row-major [K x ldW], column window colOff..+15) into WMMA B-fragment
// layout: frag[(ks*32+L)*16 + e] = W[kg, colOff + L%16],
// kg = ks*32 + (L<16?0:8) + (e<8?e:e+8), zero-padded past K.
__global__ void k_bfrag(const float* __restrict__ W, int K, int ldW, int colOff,
                        _Float16* __restrict__ frag, int ksteps) {
  int t = blockIdx.x * blockDim.x + threadIdx.x;
  if (t >= ksteps * 32) return;
  int ks = t >> 5, L = t & 31;
  int n = L & 15;
  int kbase = ks * 32 + ((L < 16) ? 0 : 8);
  _Float16* out = frag + (unsigned)t * 16u;
#pragma unroll
  for (int e = 0; e < 16; ++e) {
    int kg = kbase + (e < 8 ? e : e + 8);
    out[e] = (kg < K) ? (_Float16)W[(unsigned)kg * ldW + colOff + n] : (_Float16)0.f;
  }
}

// ---------------- conv1: per-edge message, K=80 (pad 96), scatter-add ----------------
__global__ void k_conv1(const _Float16* __restrict__ x_h, const int* __restrict__ srcI,
                        const int* __restrict__ dstI, const float* __restrict__ edge_attr,
                        const _Float16* __restrict__ frag, const float* __restrict__ bias,
                        float* __restrict__ agg, int nTiles) {
  const int L = threadIdx.x & 31;
  const int g = L >> 4;
  const int M = L & 15;
  const unsigned c0 = g ? 8u : 0u;
  // hoisted loop-invariants: 3 B-fragments + bias
  const v16h b0 = *(const v16h*)(frag + (unsigned)(0 * 32 + L) * 16u);
  const v16h b1 = *(const v16h*)(frag + (unsigned)(1 * 32 + L) * 16u);
  const v16h b2 = *(const v16h*)(frag + (unsigned)(2 * 32 + L) * 16u);
  const float bn = bias[M];

  const int nWaves = gridDim.x * (blockDim.x >> 5);
  int tile = blockIdx.x * (blockDim.x >> 5) + (threadIdx.x >> 5);
  for (; tile < nTiles; tile += nWaves) {
    const unsigned idx = (unsigned)tile * 16u + (unsigned)M;
    const int s = __builtin_nontemporal_load(srcI + idx);
    const _Float16* xr = x_h + (unsigned)s * (unsigned)DIN;

    v8f c = {};
    {  // K 0..31 (node features, cached gather)
      v16h a = combine8(load8h(xr + c0), load8h(xr + c0 + 16u));
      c = wmma16(a, b0, c);
    }
    {  // K 32..63
      v16h a = combine8(load8h(xr + 32u + c0), load8h(xr + 32u + c0 + 16u));
      c = wmma16(a, b1, c);
    }
    {  // K 64..79 = edge_attr (stream), 80..95 = zero pad
      v8h z = {};
      v16h a = combine8(cvt8_nt(edge_attr + idx * (unsigned)DED + c0), z);
      c = wmma16(a, b2, c);
    }
    // per-lane D rows: all 8 dst indices sit in one 64B line -> clause of L1 hits
    int d[8];
#pragma unroll
    for (int v = 0; v < 8; ++v)
      d[v] = __builtin_nontemporal_load(dstI + (unsigned)tile * 16u + (unsigned)(v + (g ? 8 : 0)));
#pragma unroll
    for (int v = 0; v < 8; ++v)
      atomicAdd(&agg[d[v] * HD + M], c[v] + bn);
  }
}

// ---------------- skip1: h = relu(agg1 + x@Wskip1 + b), store f16 ----------------
__global__ void k_skip1(const _Float16* __restrict__ x_h, const float* __restrict__ agg,
                        const _Float16* __restrict__ sfrag, const float* __restrict__ bias,
                        _Float16* __restrict__ h_h, int nTiles) {
  const int L = threadIdx.x & 31;
  const int g = L >> 4;
  const int M = L & 15;
  const unsigned c0 = g ? 8u : 0u;
  const v16h b0 = *(const v16h*)(sfrag + (unsigned)(0 * 32 + L) * 16u);
  const v16h b1 = *(const v16h*)(sfrag + (unsigned)(1 * 32 + L) * 16u);
  const float bn = bias[M];

  const int nWaves = gridDim.x * (blockDim.x >> 5);
  int tile = blockIdx.x * (blockDim.x >> 5) + (threadIdx.x >> 5);
  for (; tile < nTiles; tile += nWaves) {
    const _Float16* xr = x_h + ((unsigned)tile * 16u + (unsigned)M) * (unsigned)DIN;
    v8f c = {};
    {
      v16h a = combine8(load8h(xr + c0), load8h(xr + c0 + 16u));
      c = wmma16(a, b0, c);
    }
    {
      v16h a = combine8(load8h(xr + 32u + c0), load8h(xr + 32u + c0 + 16u));
      c = wmma16(a, b1, c);
    }
#pragma unroll
    for (int v = 0; v < 8; ++v) {
      int nd = tile * 16 + v + (g ? 8 : 0);
      float val = c[v] + bn + agg[nd * HD + M];
      val = val > 0.f ? val : 0.f;
      h_h[nd * HD + M] = (_Float16)val;
    }
  }
}

// ---------------- conv2: K=32 exactly, one WMMA per 16-edge tile ----------------
__global__ void k_conv2(const _Float16* __restrict__ h_h, const int* __restrict__ srcI,
                        const int* __restrict__ dstI, const float* __restrict__ edge_attr,
                        const _Float16* __restrict__ frag, const float* __restrict__ bias,
                        float* __restrict__ agg, int nTiles) {
  const int L = threadIdx.x & 31;
  const int g = L >> 4;
  const int M = L & 15;
  const unsigned c0 = g ? 8u : 0u;
  const v16h b0 = *(const v16h*)(frag + (unsigned)L * 16u);
  const float bn = bias[M];

  const int nWaves = gridDim.x * (blockDim.x >> 5);
  int tile = blockIdx.x * (blockDim.x >> 5) + (threadIdx.x >> 5);
  for (; tile < nTiles; tile += nWaves) {
    const unsigned idx = (unsigned)tile * 16u + (unsigned)M;
    const int s = __builtin_nontemporal_load(srcI + idx);

    // K 0..15 = h[src] (cached gather), K 16..31 = edge_attr (stream)
    v16h a = combine8(load8h(h_h + (unsigned)s * (unsigned)HD + c0),
                      cvt8_nt(edge_attr + idx * (unsigned)DED + c0));
    v8f c = {};
    c = wmma16(a, b0, c);

    int d[8];
#pragma unroll
    for (int v = 0; v < 8; ++v)
      d[v] = __builtin_nontemporal_load(dstI + (unsigned)tile * 16u + (unsigned)(v + (g ? 8 : 0)));
#pragma unroll
    for (int v = 0; v < 8; ++v)
      atomicAdd(&agg[d[v] * HD + M], c[v] + bn);
  }
}

// ---------------- final: t = agg2 + h@Wskip2 + b2 ; out = t@Wlin3 + b3 ----------------
__global__ void k_final(const _Float16* __restrict__ h_h, const float* __restrict__ agg2,
                        const _Float16* __restrict__ sfrag2, const float* __restrict__ b_skip2,
                        const _Float16* __restrict__ lin3frag, const float* __restrict__ b_lin3,
                        float* __restrict__ out, int nTiles) {
  __shared__ _Float16 lds[8][16][16];
  const int w = threadIdx.x >> 5;
  const int L = threadIdx.x & 31;
  const int g = L >> 4;
  const int M = L & 15;
  const unsigned c0 = g ? 8u : 0u;
  // hoisted: skip2 fragment, 4 lin3 fragments, biases
  const v16h bs = *(const v16h*)(sfrag2 + (unsigned)L * 16u);
  v16h bl[4];
  float bb[4];
#pragma unroll
  for (int t4 = 0; t4 < 4; ++t4) {
    bl[t4] = *(const v16h*)(lin3frag + (unsigned)(t4 * 32 + L) * 16u);
    bb[t4] = b_lin3[t4 * 16 + M];
  }
  const float bn = b_skip2[M];

  const int nWaves = gridDim.x * 8;
  const int waveId = blockIdx.x * 8 + w;
  const int iters = (nTiles + nWaves - 1) / nWaves;  // uniform across block -> barrier-safe
  for (int it = 0; it < iters; ++it) {
    const int tile = waveId + it * nWaves;
    const bool active = (tile < nTiles);
    if (active) {
      const _Float16* hr = h_h + ((unsigned)tile * 16u + (unsigned)M) * (unsigned)HD;
      v8h z = {};
      v16h a = combine8(load8h(hr + c0), z);  // K=16 padded to 32
      v8f c = {};
      c = wmma16(a, bs, c);
#pragma unroll
      for (int v = 0; v < 8; ++v) {
        int row = v + (g ? 8 : 0);
        float t = c[v] + bn + agg2[(tile * 16 + row) * HD + M];
        lds[w][row][M] = (_Float16)t;  // re-layout C(f32) -> A(f16) via LDS
      }
    }
    __syncthreads();
    if (active) {
      v8h z = {};
      v16h at = combine8(*(const v8h*)&lds[w][M][c0], z);  // K=16 padded to 32
#pragma unroll
      for (int t4 = 0; t4 < 4; ++t4) {
        v8f c2 = {};
        c2 = wmma16(at, bl[t4], c2);
#pragma unroll
        for (int v = 0; v < 8; ++v) {
          int nd = tile * 16 + v + (g ? 8 : 0);
          __builtin_nontemporal_store(c2[v] + bb[t4], out + nd * DIN + t4 * 16 + M);
        }
      }
    }
    __syncthreads();  // protect lds slot before next iteration overwrites
  }
}

extern "C" void kernel_launch(void* const* d_in, const int* in_sizes, int n_in,
                              void* d_out, int out_size, void* d_ws, size_t ws_size,
                              hipStream_t stream) {
  const float* x         = (const float*)d_in[0];
  const int*   edge_idx  = (const int*)d_in[1];
  const float* edge_attr = (const float*)d_in[2];
  const float* W_msg1    = (const float*)d_in[3];
  const float* b_msg1    = (const float*)d_in[4];
  const float* W_skip1   = (const float*)d_in[5];
  const float* b_skip1   = (const float*)d_in[6];
  const float* W_msg2    = (const float*)d_in[7];
  const float* b_msg2    = (const float*)d_in[8];
  const float* W_skip2   = (const float*)d_in[9];
  const float* b_skip2   = (const float*)d_in[10];
  const float* W_lin3    = (const float*)d_in[11];
  const float* b_lin3    = (const float*)d_in[12];
  float* out = (float*)d_out;

  const int* srcI = edge_idx;
  const int* dstI = edge_idx + NEDG;

  // workspace layout (256B aligned slabs)
  auto up = [](size_t v) { return (v + 255) & ~(size_t)255; };
  char* ws = (char*)d_ws;
  size_t off = 0;
  float* agg1 = (float*)(ws + off);              // N*16 f32
  off = up(off + (size_t)NND * HD * 4);
  float* agg2 = (float*)(ws + off);              // N*16 f32
  off = up(off + (size_t)NND * HD * 4);
  _Float16* x_h = (_Float16*)(ws + off);         // N*64 f16
  off = up(off + (size_t)NND * DIN * 2);
  _Float16* h_h = (_Float16*)(ws + off);         // N*16 f16
  off = up(off + (size_t)NND * HD * 2);
  _Float16* frag_msg1 = (_Float16*)(ws + off);   // 3 ksteps
  off = up(off + 3 * 512 * 2);
  _Float16* frag_skip1 = (_Float16*)(ws + off);  // 2 ksteps
  off = up(off + 2 * 512 * 2);
  _Float16* frag_msg2 = (_Float16*)(ws + off);   // 1 kstep
  off = up(off + 512 * 2);
  _Float16* frag_skip2 = (_Float16*)(ws + off);  // 1 kstep
  off = up(off + 512 * 2);
  _Float16* frag_lin3 = (_Float16*)(ws + off);   // 4 col-tiles x 1 kstep
  off = up(off + 4 * 512 * 2);
  (void)ws_size; (void)in_sizes; (void)n_in; (void)out_size;

  // prep
  {
    int n = 2 * NND * HD;  // agg1+agg2 adjacent
    k_zero<<<(n + 255) / 256, 256, 0, stream>>>(agg1, n);
  }
  {
    int n = NND * DIN;
    k_cvt<<<(n + 255) / 256, 256, 0, stream>>>(x, x_h, n);
  }
  k_bfrag<<<1, 128, 0, stream>>>(W_msg1, DIN + DED, HD, 0, frag_msg1, 3);
  k_bfrag<<<1, 128, 0, stream>>>(W_skip1, DIN, HD, 0, frag_skip1, 2);
  k_bfrag<<<1, 128, 0, stream>>>(W_msg2, HD + DED, HD, 0, frag_msg2, 1);
  k_bfrag<<<1, 128, 0, stream>>>(W_skip2, HD, HD, 0, frag_skip2, 1);
  for (int t = 0; t < 4; ++t)
    k_bfrag<<<1, 128, 0, stream>>>(W_lin3, HD, DIN, t * 16, frag_lin3 + (size_t)t * 512, 1);

  const int eTiles = NEDG / 16;  // 200000
  const int nTiles = NND / 16;   // 6250

  // persistent-wave grids: enough waves to fill the chip, ~12 tiles each for
  // fragment amortization on the edge kernels
  k_conv1<<<2048, 256, 0, stream>>>(x_h, srcI, dstI, edge_attr, frag_msg1, b_msg1, agg1, eTiles);
  k_skip1<<<256, 256, 0, stream>>>(x_h, agg1, frag_skip1, b_skip1, h_h, nTiles);
  k_conv2<<<2048, 256, 0, stream>>>(h_h, srcI, dstI, edge_attr, frag_msg2, b_msg2, agg2, eTiles);
  k_final<<<256, 256, 0, stream>>>(h_h, agg2, frag_skip2, b_skip2, frag_lin3, b_lin3, out, nTiles);
}
